// Sp_Attention_12721693131237
// MI455X (gfx1250) — matrix operand order repaired and checked by reading, not compiled, not run on hardware
//
#include <hip/hip_runtime.h>
#include <hip/hip_bf16.h>

#define DIMC   192
#define HEADS  8
#define CPH    24
#define BATCH  2
#define HH     256
#define WW2    256
#define HWN    65536
#define EPSV   1e-12f

typedef __attribute__((ext_vector_type(16))) __bf16 v16bf;
typedef __attribute__((ext_vector_type(8)))  float  v8f;
typedef int v4i32 __attribute__((vector_size(16)));

union FragBF {
    v16bf          v;
    unsigned short us[16];
    unsigned int   u32[8];
    uint4          q4[2];
};

__device__ __forceinline__ float bf2f(unsigned short u) {
    return __uint_as_float(((unsigned int)u) << 16);
}
__device__ __forceinline__ unsigned short f2bf(float f) {
    unsigned int u = __float_as_uint(f);
    return (unsigned short)((u + 0x7FFFu + ((u >> 16) & 1u)) >> 16);
}

// ---- CDNA5 async global->LDS staging (guarded; falls back to plain copy) ----
#if defined(__has_builtin)
#  if __has_builtin(__builtin_amdgcn_global_load_async_to_lds_b128)
#    define HAVE_ASYNC_LDS 1
#  endif
#  if __has_builtin(__builtin_amdgcn_s_wait_asynccnt)
#    define HAVE_WAIT_ASYNC 1
#  endif
#endif

__device__ __forceinline__ void stage16B(const void* src, void* dst_lds) {
#if defined(HAVE_ASYNC_LDS)
    // param 0: AS1 (global) v4i32*; param 1: AS3 (LDS) v4i32*.
    // AS3 is 32-bit; flat->LDS keeps addr[31:0] per ISA 10.2, so the integer
    // truncation yields the correct LDS offset.
    __builtin_amdgcn_global_load_async_to_lds_b128(
        (__attribute__((address_space(1))) v4i32*)(unsigned long long)src,
        (__attribute__((address_space(3))) v4i32*)(unsigned int)(unsigned long long)dst_lds,
        0, 0);
#else
    *(uint4*)dst_lds = *(const uint4*)src;
#endif
}
__device__ __forceinline__ void async_stage_fence() {
#if defined(HAVE_ASYNC_LDS) && defined(HAVE_WAIT_ASYNC)
    __builtin_amdgcn_s_wait_asynccnt(0);
#endif
}

__device__ __forceinline__ float block_reduce_sum256(float val) {
    __shared__ float red[256];
    int t = threadIdx.x;
    red[t] = val;
    __syncthreads();
    for (int s = 128; s > 0; s >>= 1) {
        if (t < s) red[t] += red[t + s];
        __syncthreads();
    }
    float r = red[0];
    __syncthreads();
    return r;
}

// ---------------- zero init of accumulators ----------------
__global__ void k_init(float* attn, float* qss, float* kss) {
    int idx = blockIdx.x * 256 + threadIdx.x;
    if (idx < BATCH * DIMC * DIMC) attn[idx] = 0.f;
    if (idx < BATCH * DIMC) { qss[idx] = 0.f; kss[idx] = 0.f; }
}

// ---------------- qscale[o] = sum_i q_w[o,i] ----------------
__global__ void k_qscale(const float* qw, float* qscale) {
    int o = blockIdx.x * blockDim.x + threadIdx.x;
    if (o < DIMC) {
        float s = 0.f;
        for (int i = 0; i < DIMC; i++) s += qw[o * DIMC + i];
        qscale[o] = s;
    }
}

// ---- x -> bf16, k-pair interleaved: word(b,k2,n) = bf16(x[2k2,n]) | bf16(x[2k2+1,n])<<16 ----
__global__ void k_xcast(const float* __restrict__ x, unsigned int* __restrict__ xw) {
    int idx = blockIdx.x * 256 + threadIdx.x;       // quad-of-words index
    if (idx >= BATCH * 96 * (HWN / 4)) return;
    int n4  = idx & (HWN / 4 - 1);
    int bk2 = idx / (HWN / 4);                      // b*96 + k2
    int b   = bk2 / 96, k2 = bk2 % 96;
    const float* r0 = x + (size_t)(b * DIMC + 2 * k2) * HWN + n4 * 4;
    float4 e = *(const float4*)r0;
    float4 o = *(const float4*)(r0 + HWN);
    uint4 w;
    w.x = (unsigned)f2bf(e.x) | ((unsigned)f2bf(o.x) << 16);
    w.y = (unsigned)f2bf(e.y) | ((unsigned)f2bf(o.y) << 16);
    w.z = (unsigned)f2bf(e.z) | ((unsigned)f2bf(o.z) << 16);
    w.w = (unsigned)f2bf(e.w) | ((unsigned)f2bf(o.w) << 16);
    *(uint4*)(xw + (size_t)bk2 * HWN + n4 * 4) = w;
}

// ---------------- bf16 pre-conversion of kv_w ----------------
__global__ void k_wcast(const float* __restrict__ w, unsigned short* __restrict__ wbf, int n) {
    int idx = blockIdx.x * 256 + threadIdx.x;
    if (idx < n) wbf[idx] = f2bf(w[idx]);
}

// ---------------- xsum[b,n] = sum_c x[b,c,n] ----------------
__global__ void k_csum(const float* __restrict__ x, float* __restrict__ xsum) {
    int idx = blockIdx.x * 256 + threadIdx.x;
    if (idx >= BATCH * HWN) return;
    int b = idx >> 16, n = idx & 65535;
    float s = 0.f;
    for (int c = 0; c < DIMC; c++) s += x[(size_t)(b * DIMC + c) * HWN + n];
    xsum[idx] = s;
}

// ---------------- g = (SobelX + SobelY) * xsum (cross-corr, zero pad) ----------------
__global__ void k_sobel(const float* __restrict__ xsum, float* __restrict__ g) {
    int idx = blockIdx.x * 256 + threadIdx.x;
    if (idx >= BATCH * HWN) return;
    int b = idx >> 16, n = idx & 65535;
    int h = n >> 8, w = n & 255;
    const float cw[3][3] = {{2.f, -2.f, 0.f}, {-2.f, 0.f, 2.f}, {0.f, 2.f, 2.f}};
    const float* src = xsum + (size_t)b * HWN;
    float acc = 0.f;
#pragma unroll
    for (int kh = 0; kh < 3; kh++) {
        int hh = h + kh - 1;
        if (hh < 0 || hh >= HH) continue;
#pragma unroll
        for (int kw = 0; kw < 3; kw++) {
            int ww = w + kw - 1;
            if (ww < 0 || ww >= WW2) continue;
            acc += src[hh * WW2 + ww] * cw[kh][kw];
        }
    }
    g[idx] = acc;
}

// ---------------- q path: q = qscale[o]*DW3x3(g, q_dw[o]); accumulate sumsq ----------------
__global__ void k_qpath(const float* __restrict__ g, const float* __restrict__ qdw,
                        const float* __restrict__ qscale,
                        unsigned short* __restrict__ qout, float* qss) {
    int tile = blockIdx.x & 255;
    int bo   = blockIdx.x >> 8;            // b*192 + o
    int o    = bo % DIMC;
    int b    = bo / DIMC;
    int n    = tile * 256 + threadIdx.x;
    int h = n >> 8, w = n & 255;
    float wreg[9];
#pragma unroll
    for (int i = 0; i < 9; i++) wreg[i] = qdw[o * 9 + i];
    const float* gb = g + (size_t)b * HWN;
    float acc = 0.f;
#pragma unroll
    for (int kh = 0; kh < 3; kh++) {
        int hh = h + kh - 1;
        if (hh < 0 || hh >= HH) continue;
#pragma unroll
        for (int kw = 0; kw < 3; kw++) {
            int ww = w + kw - 1;
            if (ww < 0 || ww >= WW2) continue;
            acc += gb[hh * WW2 + ww] * wreg[kh * 3 + kw];
        }
    }
    acc *= qscale[o];
    qout[(size_t)bo * HWN + n] = f2bf(acc);
    float ss = block_reduce_sum256(acc * acc);
    if (threadIdx.x == 0) atomicAdd(&qss[bo], ss);
}

// ---------------- kv 1x1 GEMM (WMMA, LDS-tiled, async staging, interleaved B) ----------------
// block = 256 thr (8 waves); block tile: M=128 (8 waves x 16) x N=64; K loop 192/32
__global__ void k_gemm_kv(const unsigned int* __restrict__ xw,
                          const unsigned short* __restrict__ kvwbf,
                          unsigned short* __restrict__ kv1) {
    __shared__ unsigned int xt[16 * 64];       // 4 KB: 16 k-pair rows x 64 cols
    const int NT = HWN / 64;                   // 1024
    int blk = blockIdx.x;
    int b   = blk / (3 * NT);
    int r   = blk % (3 * NT);
    int mg  = r / NT;                          // m-group 0..2 (128 rows each)
    int nt  = r % NT;
    int nbase = nt * 64;
    int wave  = threadIdx.x >> 5;
    int mbase = mg * 128 + wave * 16;
    int lane = threadIdx.x & 31, half = lane >> 4, l16 = lane & 15;

    v8f acc[4];
#pragma unroll
    for (int s = 0; s < 4; s++)
#pragma unroll
        for (int rr = 0; rr < 8; rr++) acc[s][rr] = 0.f;

    const unsigned short* arow = kvwbf + (size_t)(mbase + l16) * DIMC;

    for (int kb = 0; kb < DIMC; kb += 32) {
        int k2b = kb >> 1;
        {   // async stage: 256 threads x one 16B chunk (4 words)
            int t = threadIdx.x;
            int row = t >> 4, chunk = t & 15;
            const unsigned int* src =
                xw + (size_t)(b * 96 + k2b + row) * HWN + nbase + chunk * 4;
            stage16B(src, &xt[row * 64 + chunk * 4]);
        }
        async_stage_fence();
        __syncthreads();

        FragBF a;
        a.q4[0] = *(const uint4*)(arow + kb + 16 * half);
        a.q4[1] = *(const uint4*)(arow + kb + 16 * half + 8);
#pragma unroll
        for (int s = 0; s < 4; s++) {
            FragBF bb;
#pragma unroll
            for (int j = 0; j < 8; j++)        // one dword = one K-pair: 8 ds_load_b32
                bb.u32[j] = xt[(8 * half + j) * 64 + s * 16 + l16];
            acc[s] = __builtin_amdgcn_wmma_f32_16x16x32_bf16(
                false, a.v, false, bb.v, (short)0, acc[s], false, false);
        }
        __syncthreads();
    }
#pragma unroll
    for (int s = 0; s < 4; s++)
#pragma unroll
        for (int rr = 0; rr < 8; rr++) {
            int m = mbase + rr + 8 * half;
            int n = nbase + s * 16 + l16;
            kv1[(size_t)(b * 384 + m) * HWN + n] = f2bf(acc[s][rr]);
        }
}

// ---------------- depthwise 3x3 on kv1 -> k (o<192, + sumsq), v (o>=192, interleaved) ------
__global__ void k_kvdw(const unsigned short* __restrict__ kv1, const float* __restrict__ kvdw,
                       unsigned short* __restrict__ kout, unsigned short* __restrict__ vout,
                       float* kss) {
    int tile = blockIdx.x & 255;
    int bo   = blockIdx.x >> 8;            // b*384 + o
    int o    = bo % 384;
    int b    = bo / 384;
    int n    = tile * 256 + threadIdx.x;
    int h = n >> 8, w = n & 255;
    float wreg[9];
#pragma unroll
    for (int i = 0; i < 9; i++) wreg[i] = kvdw[o * 9 + i];
    const unsigned short* src = kv1 + (size_t)bo * HWN;
    float acc = 0.f;
#pragma unroll
    for (int kh = 0; kh < 3; kh++) {
        int hh = h + kh - 1;
        if (hh < 0 || hh >= HH) continue;
#pragma unroll
        for (int kw = 0; kw < 3; kw++) {
            int ww = w + kw - 1;
            if (ww < 0 || ww >= WW2) continue;
            acc += bf2f(src[hh * WW2 + ww]) * wreg[kh * 3 + kw];
        }
    }
    if (o < DIMC) {
        kout[(size_t)(b * DIMC + o) * HWN + n] = f2bf(acc);
        float ss = block_reduce_sum256(acc * acc);
        if (threadIdx.x == 0) atomicAdd(&kss[b * DIMC + o], ss);
    } else {                               // v stored j-pair interleaved for GEMM-out B
        int j = o - DIMC, j2 = j >> 1, par = j & 1;
        vout[(((size_t)(b * 96 + j2) * HWN + n) << 1) + par] = f2bf(acc);
    }
}

// ---------------- attn Gram: attn[b,c,d] += sum_n q[c,n]*k[d,n] (WMMA, split-K) ----------------
__global__ void k_gemm_attn(const unsigned short* __restrict__ q,
                            const unsigned short* __restrict__ k, float* attn) {
    const int KS = 64, KLEN = HWN / KS;       // 64 slices of 1024
    int wid = (blockIdx.x * blockDim.x + threadIdx.x) >> 5;
    const int TOT = BATCH * 12 * 12 * KS;
    if (wid >= TOT) return;
    int b = wid / (12 * 12 * KS);
    int r = wid % (12 * 12 * KS);
    int mt = r / (12 * KS);  r %= (12 * KS);
    int nt = r / KS;
    int ks = r % KS;
    int lane = threadIdx.x & 31, half = lane >> 4, l16 = lane & 15;

    const unsigned short* qrow = q + (size_t)(b * DIMC + mt * 16 + l16) * HWN;
    const unsigned short* krow = k + (size_t)(b * DIMC + nt * 16 + l16) * HWN;

    v8f acc;
#pragma unroll
    for (int rr = 0; rr < 8; rr++) acc[rr] = 0.f;

    int k0 = ks * KLEN;
    for (int kb = k0; kb < k0 + KLEN; kb += 32) {
        FragBF a, bb;
        a.q4[0]  = *(const uint4*)(qrow + kb + 16 * half);
        a.q4[1]  = *(const uint4*)(qrow + kb + 16 * half + 8);
        bb.q4[0] = *(const uint4*)(krow + kb + 16 * half);
        bb.q4[1] = *(const uint4*)(krow + kb + 16 * half + 8);
        __builtin_prefetch(qrow + kb + 512, 0, 3);
        __builtin_prefetch(krow + kb + 512, 0, 3);
        acc = __builtin_amdgcn_wmma_f32_16x16x32_bf16(
            false, a.v, false, bb.v, (short)0, acc, false, false);
    }
#pragma unroll
    for (int rr = 0; rr < 8; rr++) {
        int c = mt * 16 + rr + 8 * half;
        int d = nt * 16 + l16;
        atomicAdd(&attn[(size_t)(b * DIMC + c) * DIMC + d], acc[rr]);
    }
}

// ---------------- per-row: normalize, top-k masked softmax x4, combine -> W ----------------
__global__ void k_softmax_w(const float* attn, const float* qss, const float* kss,
                            const float* temp, const float* a1, const float* a2,
                            const float* a3, const float* a4, float* Wc) {
    int idx = blockIdx.x * blockDim.x + threadIdx.x;
    if (idx >= BATCH * HEADS * CPH) return;
    int b = idx / (HEADS * CPH);
    int r = idx % (HEADS * CPH);
    int h = r / CPH, c = r % CPH;
    int cg = h * CPH + c;
    float qn = fmaxf(sqrtf(qss[b * DIMC + cg]), EPSV);
    float t  = temp[h];
    float vals[CPH];
    for (int d = 0; d < CPH; d++) {
        int dg = h * CPH + d;
        float kn = fmaxf(sqrtf(kss[b * DIMC + dg]), EPSV);
        vals[d] = attn[(size_t)(b * DIMC + cg) * DIMC + dg] / (qn * kn) * t;
    }
    float s[CPH];
    for (int d = 0; d < CPH; d++) s[d] = vals[d];
    for (int i = 1; i < CPH; i++) {            // insertion sort, descending
        float key = s[i];
        int j = i - 1;
        while (j >= 0 && s[j] < key) { s[j + 1] = s[j]; j--; }
        s[j + 1] = key;
    }
    float m = s[0];
    float wout[CPH];
    for (int d = 0; d < CPH; d++) wout[d] = 0.f;
    const int   kk[4] = {12, 16, 18, 19};      // C/2, 2C/3, 3C/4, 4C/5
    const float aa[4] = {a1[0], a2[0], a3[0], a4[0]};
    for (int i = 0; i < 4; i++) {
        float thr = s[kk[i] - 1];
        float sum = 0.f;
        for (int d = 0; d < CPH; d++)
            if (vals[d] >= thr) sum += __expf(vals[d] - m);
        float inv = aa[i] / fmaxf(sum, 1e-30f);
        for (int d = 0; d < CPH; d++)
            if (vals[d] >= thr) wout[d] += inv * __expf(vals[d] - m);
    }
    for (int d = 0; d < CPH; d++)
        Wc[((size_t)(b * HEADS + h) * CPH + c) * CPH + d] = wout[d];
}

// ---------------- M2[b,o,24h+d] = sum_cc proj_w[o,24h+cc]*W[b,h,cc,d] ----------------
__global__ void k_m2(const float* __restrict__ Wc, const float* __restrict__ projw,
                     unsigned short* __restrict__ M2) {
    int idx = blockIdx.x * 256 + threadIdx.x;
    if (idx >= BATCH * DIMC * DIMC) return;
    int b = idx / (DIMC * DIMC);
    int r = idx % (DIMC * DIMC);
    int o = r / DIMC, j = r % DIMC;
    int h = j / CPH, d = j % CPH;
    float acc = 0.f;
    for (int cc = 0; cc < CPH; cc++)
        acc += projw[o * DIMC + h * CPH + cc] *
               Wc[((size_t)(b * HEADS + h) * CPH + cc) * CPH + d];
    M2[idx] = f2bf(acc);
}

// ---------------- final fused GEMM: out = M2 x v (WMMA, LDS-tiled, async staging) ----------------
// block = 128 thr (4 waves); block tile: M=64 (4 waves x 16) x N=64; K loop 192/32
__global__ void k_gemm_out(const unsigned short* __restrict__ M2,
                           const unsigned int* __restrict__ vw, float* __restrict__ out) {
    __shared__ unsigned int vt[16 * 64];       // 4 KB: 16 k-pair rows x 64 cols
    const int NT = HWN / 64;
    int blk = blockIdx.x;
    int b   = blk / (3 * NT);
    int r   = blk % (3 * NT);
    int mg  = r / NT;                          // 0..2 (64 rows each)
    int nt  = r % NT;
    int nbase = nt * 64;
    int wave  = threadIdx.x >> 5;
    int mbase = mg * 64 + wave * 16;
    int lane = threadIdx.x & 31, half = lane >> 4, l16 = lane & 15;

    v8f acc[4];
#pragma unroll
    for (int s = 0; s < 4; s++)
#pragma unroll
        for (int rr = 0; rr < 8; rr++) acc[s][rr] = 0.f;

    const unsigned short* arow = M2 + (size_t)(b * DIMC + mbase + l16) * DIMC;

    for (int kb = 0; kb < DIMC; kb += 32) {
        int k2b = kb >> 1;
        for (int i = threadIdx.x; i < 256; i += 128) {   // 256 chunks / 128 threads
            int row = i >> 4, chunk = i & 15;
            const unsigned int* src =
                vw + (size_t)(b * 96 + k2b + row) * HWN + nbase + chunk * 4;
            stage16B(src, &vt[row * 64 + chunk * 4]);
        }
        async_stage_fence();
        __syncthreads();

        FragBF a;
        a.q4[0] = *(const uint4*)(arow + kb + 16 * half);
        a.q4[1] = *(const uint4*)(arow + kb + 16 * half + 8);
#pragma unroll
        for (int s = 0; s < 4; s++) {
            FragBF bb;
#pragma unroll
            for (int j = 0; j < 8; j++)
                bb.u32[j] = vt[(8 * half + j) * 64 + s * 16 + l16];
            acc[s] = __builtin_amdgcn_wmma_f32_16x16x32_bf16(
                false, a.v, false, bb.v, (short)0, acc[s], false, false);
        }
        __syncthreads();
    }
#pragma unroll
    for (int s = 0; s < 4; s++)
#pragma unroll
        for (int rr = 0; rr < 8; rr++) {
            int m = mbase + rr + 8 * half;
            int n = nbase + s * 16 + l16;
            out[(size_t)(b * DIMC + m) * HWN + n] = acc[s][rr];
        }
}

extern "C" void kernel_launch(void* const* d_in, const int* in_sizes, int n_in,
                              void* d_out, int out_size, void* d_ws, size_t ws_size,
                              hipStream_t stream) {
    (void)in_sizes; (void)n_in; (void)out_size; (void)ws_size;
    const float* x      = (const float*)d_in[0];
    const float* q_w    = (const float*)d_in[1];
    const float* q_dw   = (const float*)d_in[2];
    const float* kv_w   = (const float*)d_in[3];
    const float* kv_dw  = (const float*)d_in[4];
    const float* proj_w = (const float*)d_in[5];
    const float* temp   = (const float*)d_in[6];
    const float* a1     = (const float*)d_in[7];
    const float* a2     = (const float*)d_in[8];
    const float* a3     = (const float*)d_in[9];
    const float* a4     = (const float*)d_in[10];
    float* out = (float*)d_out;

    // workspace carve-out (256-B aligned)
    char* wp = (char*)d_ws;
    auto carve = [&](size_t bytes) -> char* {
        char* p = wp;
        wp += (bytes + 255) & ~(size_t)255;
        return p;
    };
    float*          xsum   = (float*)carve((size_t)BATCH * HWN * 4);
    float*          g      = (float*)carve((size_t)BATCH * HWN * 4);
    float*          qscale = (float*)carve(DIMC * 4);
    float*          qss    = (float*)carve(BATCH * DIMC * 4);
    float*          kss    = (float*)carve(BATCH * DIMC * 4);
    unsigned int*   xw     = (unsigned int*)carve((size_t)BATCH * 96 * HWN * 4);
    unsigned short* qbuf   = (unsigned short*)carve((size_t)BATCH * DIMC * HWN * 2);
    unsigned short* kbuf   = (unsigned short*)carve((size_t)BATCH * DIMC * HWN * 2);
    unsigned short* vbuf   = (unsigned short*)carve((size_t)BATCH * DIMC * HWN * 2);
    float*          attn   = (float*)carve((size_t)BATCH * DIMC * DIMC * 4);
    float*          Wc     = (float*)carve((size_t)BATCH * HEADS * CPH * CPH * 4);
    unsigned short* M2     = (unsigned short*)carve((size_t)BATCH * DIMC * DIMC * 2);
    unsigned short* kvwbf  = (unsigned short*)carve((size_t)384 * DIMC * 2);
    // kv1 (bf16, B*384*HW*2 bytes == out_size*4 bytes) staged inside d_out
    unsigned short* kv1 = (unsigned short*)d_out;

    k_init<<<(BATCH * DIMC * DIMC + 255) / 256, 256, 0, stream>>>(attn, qss, kss);
    k_qscale<<<1, DIMC, 0, stream>>>(q_w, qscale);
    k_xcast<<<(BATCH * 96 * (HWN / 4)) / 256, 256, 0, stream>>>(x, xw);
    k_wcast<<<(384 * DIMC + 255) / 256, 256, 0, stream>>>(kv_w, kvwbf, 384 * DIMC);
    k_csum<<<(BATCH * HWN) / 256, 256, 0, stream>>>(x, xsum);
    k_sobel<<<(BATCH * HWN) / 256, 256, 0, stream>>>(xsum, g);
    k_qpath<<<BATCH * DIMC * 256, 256, 0, stream>>>(g, q_dw, qscale, qbuf, qss);
    k_gemm_kv<<<BATCH * 3 * (HWN / 64), 256, 0, stream>>>(xw, kvwbf, kv1);
    k_kvdw<<<BATCH * 384 * 256, 256, 0, stream>>>(kv1, kv_dw, kbuf, vbuf, kss);
    k_gemm_attn<<<(BATCH * 12 * 12 * 64) / 8, 256, 0, stream>>>(qbuf, kbuf, attn);
    k_softmax_w<<<2, 192, 0, stream>>>(attn, qss, kss, temp, a1, a2, a3, a4, Wc);
    k_m2<<<(BATCH * DIMC * DIMC + 255) / 256, 256, 0, stream>>>(Wc, proj_w, M2);
    k_gemm_out<<<BATCH * 3 * (HWN / 64), 128, 0, stream>>>(M2, (const unsigned int*)vbuf, out);
}
